// approach_module_12489764897243
// MI455X (gfx1250) — matrix operand order repaired and checked by reading, compiled, not run on hardware
//
#include <hip/hip_runtime.h>

// ---------------------------------------------------------------------------
// CDNA5 (gfx1250) fused multi-tower MLP, wave32 + v_wmma_f32_16x16x32_f16.
// ~524MB input @ 23.3TB/s => ~23us HBM floor. Each wave owns 32 rows (two
// 16-row M-tiles) so every B fragment load feeds two WMMAs. LayerNorm is
// computed in A-fragment layout (1 shfl per stat), gamma/beta pre-packed in
// A-lane order. B fragments for a whole output column are hoisted into an
// explicit array so the loads form one clause instead of a serialized
// load->wait(0)->wmma chain.
// ---------------------------------------------------------------------------

typedef __attribute__((ext_vector_type(16))) _Float16 v16h;
typedef __attribute__((ext_vector_type(8)))  _Float16 v8h;
typedef __attribute__((ext_vector_type(8)))  float    v8f;

#define LDS_STRIDE 72                 // halves per row in the transpose slab
#define SLAB_H     (16 * LDS_STRIDE)  // one 16-row sub-tile slab
#define LDS_FENCE() __threadfence_block()

struct TailConsts {
    const float *kb2, *vb2, *qb2, *ab2, *rb3, *gb2;
};

__device__ __forceinline__ v8f vzero8() {
    v8f z;
#pragma unroll
    for (int i = 0; i < 8; ++i) z[i] = 0.0f;
    return z;
}

__device__ __forceinline__ v8f wmma16(v16h a, v16h b, v8f c) {
    return __builtin_amdgcn_wmma_f32_16x16x32_f16(
        false, a, false, b, (short)0, c, false, false);
}

// Packed B fragment: 16 contiguous halves per (tile, lane) = one 32B load.
__device__ __forceinline__ v16h load_bfrag(const _Float16* wp, int tile, int lane) {
    return *(const v16h*)(wp + (((size_t)(tile * 32 + lane)) << 4));
}

// A fragment (16x32) from a global f32 row; p = row start + kt*32.
__device__ __forceinline__ v16h afrag_rep(const float* p, int c0) {
    float4 x0 = *(const float4*)(p + c0);
    float4 x1 = *(const float4*)(p + c0 + 4);
    float4 y0 = *(const float4*)(p + c0 + 16);
    float4 y1 = *(const float4*)(p + c0 + 20);
    v16h a;
    a[0]  = (_Float16)x0.x; a[1]  = (_Float16)x0.y; a[2]  = (_Float16)x0.z; a[3]  = (_Float16)x0.w;
    a[4]  = (_Float16)x1.x; a[5]  = (_Float16)x1.y; a[6]  = (_Float16)x1.z; a[7]  = (_Float16)x1.w;
    a[8]  = (_Float16)y0.x; a[9]  = (_Float16)y0.y; a[10] = (_Float16)y0.z; a[11] = (_Float16)y0.w;
    a[12] = (_Float16)y1.x; a[13] = (_Float16)y1.y; a[14] = (_Float16)y1.z; a[15] = (_Float16)y1.w;
    return a;
}

// A fragment whose K 0..2 holds the 3 per-row parameters, rest zero.
__device__ __forceinline__ v16h afrag_par(const float* par, int rowBase, int lane) {
    v16h a;
#pragma unroll
    for (int i = 0; i < 16; ++i) a[i] = (_Float16)0.0f;
    if (lane < 16) {
        const float* p = par + (size_t)(rowBase + lane) * 3;
        a[0] = (_Float16)p[0];
        a[1] = (_Float16)p[1];
        a[2] = (_Float16)p[2];
    }
    return a;
}

// Rebuild an A fragment from the LDS slab (two aligned ds_load_b128).
__device__ __forceinline__ v16h afrag_lds(const _Float16* hbuf, int kt, int lane) {
    const _Float16* p = hbuf + (lane & 15) * LDS_STRIDE + kt * 32 + ((lane >> 4) << 3);
    v8h lo = *(const v8h*)p;
    v8h hi = *(const v8h*)(p + 16);
    v16h a;
#pragma unroll
    for (int i = 0; i < 8; ++i) { a[i] = lo[i]; a[i + 8] = hi[i]; }
    return a;
}

// Store one 16x16 f32 D tile as f16 into the LDS slab at column offset colOff.
__device__ __forceinline__ void store_tile_h(const v8f& v, _Float16* hbuf, int colOff, int lane) {
    const int col = lane & 15;
    const int hi8 = (lane >> 4) << 3;
#pragma unroll
    for (int m = 0; m < 8; ++m)
        hbuf[(m + hi8) * LDS_STRIDE + colOff + col] = (_Float16)v[m];
}

// Row mean/rstd in A layout: lane holds half its row, lane^16 the other half.
template <int NFRAG, int EPF>
__device__ __forceinline__ void row_stats(const v16h* f, float invw, float& mean, float& rstd) {
    float s = 0.0f, s2 = 0.0f;
#pragma unroll
    for (int i = 0; i < NFRAG; ++i) {
#pragma unroll
        for (int e = 0; e < EPF; ++e) {
            float x = (float)f[i][e];
            s += x;
            s2 = fmaf(x, x, s2);
        }
    }
    s  += __shfl_xor(s, 16);
    s2 += __shfl_xor(s2, 16);
    mean = s * invw;
    float var = fmaxf(s2 * invw - mean * mean, 0.0f);
    rstd = rsqrtf(var + 1e-5f);
}

// (alpha*x+beta)*gamma + beta2, ReLU, applied in-place to A fragments.
// gamma/beta2 are pre-packed in A-fragment lane order (zero in padded K).
template <int NFRAG>
__device__ __forceinline__ void norm_relu(v16h* f, const _Float16* gp, const _Float16* bp,
                                          float mean, float rstd, int lane) {
    const _Float16 ah = (_Float16)rstd;
    const _Float16 bh = (_Float16)(-mean * rstd);
#pragma unroll
    for (int i = 0; i < NFRAG; ++i) {
        v16h g = *(const v16h*)(gp + (((size_t)(i * 32 + lane)) << 4));
        v16h b = *(const v16h*)(bp + (((size_t)(i * 32 + lane)) << 4));
#pragma unroll
        for (int e = 0; e < 16; ++e) {
            _Float16 t = f[i][e] * ah + bh;
            t = t * g[e] + b[e];
            f[i][e] = (t > (_Float16)0.0f) ? t : (_Float16)0.0f;
        }
    }
}

// D tiles (raw) -> LDS -> A fragments -> LN+ReLU in A layout, for both mi.
template <int T, int W>
__device__ __forceinline__ void ln_transpose(const v8f (*acc)[4], _Float16* hbuf,
                                             const _Float16* gp, const _Float16* bp,
                                             int lane, v16h (*fr)[2]) {
    constexpr int NFRAG = (T + 1) / 2;
    constexpr int EPF   = (W >= 32) ? 16 : 8;
    const float invw = 1.0f / (float)W;
#pragma unroll
    for (int mi = 0; mi < 2; ++mi)
#pragma unroll
        for (int t = 0; t < T; ++t)
            store_tile_h(acc[mi][t], hbuf + mi * SLAB_H, t * 16, lane);
    LDS_FENCE();
#pragma unroll
    for (int mi = 0; mi < 2; ++mi) {
#pragma unroll
        for (int i = 0; i < NFRAG; ++i)
            fr[mi][i] = afrag_lds(hbuf + mi * SLAB_H, i, lane);
        float mean, rstd;
        row_stats<NFRAG, EPF>(fr[mi], invw, mean, rstd);
        norm_relu<NFRAG>(fr[mi], gp, bp, mean, rstd, lane);
    }
    LDS_FENCE();
}

// Stage-1 GEMM (Kin -> 64) for both M-tiles: all KT B fragments for one
// output column are loaded up-front so the loads form one clause.
template <int KT>
__device__ __forceinline__ void gemm_stage1(const v16h (*aK)[4], const v16h* aP,
                                            const _Float16* Wp, v8f (*acc)[4], int lane) {
#pragma unroll
    for (int nt = 0; nt < 4; ++nt) {
        v16h b[KT];
#pragma unroll
        for (int kt = 0; kt < KT; ++kt) b[kt] = load_bfrag(Wp, kt * 4 + nt, lane);
        v8f a0 = vzero8(), a1 = vzero8();
#pragma unroll
        for (int kt = 0; kt < KT && kt < 4; ++kt) {
            a0 = wmma16(aK[0][kt], b[kt], a0);
            a1 = wmma16(aK[1][kt], b[kt], a1);
        }
        if (KT > 4) {  // params tail (residual tower), constant-folded
            a0 = wmma16(aP[0], b[KT - 1], a0);
            a1 = wmma16(aP[1], b[KT - 1], a1);
        }
        acc[0][nt] = a0; acc[1][nt] = a1;
    }
}

// Stage-2 GEMM (64 -> 32) for both M-tiles; all 4 B tiles preloaded.
__device__ __forceinline__ void gemm_stage2(const v16h (*fr)[2], const _Float16* Wp,
                                            const float* bias, v8f (*res)[2],
                                            int lane, int col) {
    v16h b[4];
#pragma unroll
    for (int i = 0; i < 4; ++i) b[i] = load_bfrag(Wp, i, lane);  // tile = kt*2+nt
#pragma unroll
    for (int nt = 0; nt < 2; ++nt) {
        float bb = bias ? bias[nt * 16 + col] : 0.0f;
#pragma unroll
        for (int mi = 0; mi < 2; ++mi) {
            v8f a = vzero8();
            a = wmma16(fr[mi][0], b[nt], a);
            a = wmma16(fr[mi][1], b[2 + nt], a);
#pragma unroll
            for (int m = 0; m < 8; ++m) a[m] += bb;
            res[mi][nt] = a;
        }
    }
}

// ---------------------------------------------------------------------------
// Weight pre-pack: f32 (K x N) -> f16 B-operand 32x16 tiles, zero padded.
// ---------------------------------------------------------------------------
__global__ __launch_bounds__(256) void pack_w_kernel(const float* __restrict__ W,
                                                     _Float16* __restrict__ dst,
                                                     int Kdim, int Ndim, int Nt, int total) {
    int idx = blockIdx.x * 256 + threadIdx.x;
    if (idx >= total) return;
    int e    = idx & 15;
    int lane = (idx >> 4) & 31;
    int tile = idx >> 9;
    int kt = tile / Nt, nt = tile - kt * Nt;
    int k = kt * 32 + ((e < 8) ? (((lane >> 4) << 3) + e) : (16 + ((lane >> 4) << 3) + (e - 8)));
    int n = nt * 16 + (lane & 15);
    float v = (k < Kdim && n < Ndim) ? W[k * Ndim + n] : 0.0f;
    dst[idx] = (_Float16)v;
}

// Pack a per-K vector (gamma/beta) into A-fragment lane order, zero padded.
__global__ __launch_bounds__(256) void pack_vec_kernel(const float* __restrict__ src,
                                                       _Float16* __restrict__ dst,
                                                       int dim, int total) {
    int idx = blockIdx.x * 256 + threadIdx.x;
    if (idx >= total) return;
    int e    = idx & 15;
    int lane = (idx >> 4) & 31;
    int kt   = idx >> 9;
    int c0   = (lane >> 4) << 3;
    int k    = kt * 32 + ((e < 8) ? (c0 + e) : (16 + c0 + (e - 8)));
    dst[idx] = (_Float16)((k < dim) ? src[k] : 0.0f);
}

// ---------------------------------------------------------------------------
// Main fused kernel: 256 threads = 8 waves; each wave owns 32 rows (mi=0,1).
// ---------------------------------------------------------------------------
__global__ __launch_bounds__(256, 1) void fused_towers_kernel(
    const float* __restrict__ rep, const float* __restrict__ par,
    const _Float16* __restrict__ wp, TailConsts cst, float* __restrict__ out) {

    __shared__ _Float16 smem[8 * 2 * SLAB_H];
    const int lane = threadIdx.x & 31;
    const int wave = threadIdx.x >> 5;
    _Float16* hbuf = smem + wave * 2 * SLAB_H;
    const int rowBase = blockIdx.x * 256 + wave * 32;
    const int rr  = lane & 15;
    const int c0  = (lane >> 4) << 3;
    const int col = rr;
    const int hi8 = c0;

    // Warm WGP$ with the packed weight+vector region (global_prefetch_b8).
    {
        const char* wb = (const char*)wp;
#pragma unroll
        for (int i = 0; i < 3; ++i)
            __builtin_prefetch(wb + ((size_t)threadIdx.x + (size_t)i * 256) * 128, 0, 3);
    }

    // packed weight / vector offsets (halves)
    const _Float16* kW1p = wp +     0;
    const _Float16* vW1p = wp +  8192;
    const _Float16* rW1p = wp + 16384;
    const _Float16* kW2p = wp + 26624;
    const _Float16* vW2p = wp + 28672;
    const _Float16* rW2p = wp + 30720;
    const _Float16* qW1p = wp + 32768;
    const _Float16* qW2p = wp + 33280;
    const _Float16* aW1p = wp + 34304;
    const _Float16* aW2p = wp + 35328;
    const _Float16* rW3p = wp + 35840;
    const _Float16* gW1p = wp + 36352;
    const _Float16* gW2p = wp + 36864;
    const _Float16* kg1p = wp + 37376;
    const _Float16* kb1p = wp + 38400;
    const _Float16* vg1p = wp + 39424;
    const _Float16* vb1p = wp + 40448;
    const _Float16* rg1p = wp + 41472;
    const _Float16* rb1p = wp + 42496;
    const _Float16* rg2p = wp + 43520;
    const _Float16* rb2p = wp + 44032;
    const _Float16* ag1p = wp + 44544;
    const _Float16* ab1p = wp + 45056;
    const _Float16* qg1p = wp + 45568;
    const _Float16* qb1p = wp + 46080;
    const _Float16* gg1p = wp + 46592;
    const _Float16* gb1p = wp + 47104;

    // A fragments for both 16-row sub-tiles (each row's 512B read once)
    v16h aK[2][4];
    v16h aP[2];
#pragma unroll
    for (int mi = 0; mi < 2; ++mi) {
        const float* rowp = rep + (size_t)(rowBase + mi * 16 + rr) * 128;
#pragma unroll
        for (int kt = 0; kt < 4; ++kt) aK[mi][kt] = afrag_rep(rowp + kt * 32, c0);
        aP[mi] = afrag_par(par, rowBase + mi * 16, lane);
    }

    v8f acc[2][4];
    v16h fr[2][2];

    // ========== residual tower: 131 -> 64 (LN,ReLU) -> 32 ==========
    gemm_stage1<5>(aK, aP, rW1p, acc, lane);
    ln_transpose<4, 64>(acc, hbuf, rg1p, rb1p, lane, fr);
    v8f r2[2][2];
    gemm_stage2(fr, rW2p, nullptr, r2, lane, col);   // no bias before second LN

    // ========== key tower: 128 -> 64 (LN,ReLU) -> 32 ==========
    gemm_stage1<4>(aK, aP, kW1p, acc, lane);
    ln_transpose<4, 64>(acc, hbuf, kg1p, kb1p, lane, fr);
    v8f k2[2][2];
    gemm_stage2(fr, kW2p, cst.kb2, k2, lane, col);

    // ========== value tower: 128 -> 64 (LN,ReLU) -> 32 ==========
    gemm_stage1<4>(aK, aP, vW1p, acc, lane);
    ln_transpose<4, 64>(acc, hbuf, vg1p, vb1p, lane, fr);
    v8f v2[2][2];
    gemm_stage2(fr, vW2p, cst.vb2, v2, lane, col);

    // ========== query tower: 3 -> 16 (LN,ReLU) -> 32 ==========
    {
        v16h b = load_bfrag(qW1p, 0, lane);
#pragma unroll
        for (int mi = 0; mi < 2; ++mi) acc[mi][0] = wmma16(aP[mi], b, vzero8());
    }
    ln_transpose<1, 16>(acc, hbuf, qg1p, qb1p, lane, fr);
    v8f q2[2][2];
    {
        v16h b[2];
        b[0] = load_bfrag(qW2p, 0, lane);
        b[1] = load_bfrag(qW2p, 1, lane);
#pragma unroll
        for (int nt = 0; nt < 2; ++nt) {
            float bb = cst.qb2[nt * 16 + col];
#pragma unroll
            for (int mi = 0; mi < 2; ++mi) {
                v8f a = wmma16(fr[mi][0], b[nt], vzero8());
#pragma unroll
                for (int m = 0; m < 8; ++m) a[m] += bb;
                q2[mi][nt] = a;
            }
        }
    }

    // ========== attention mix: softmax(q*k over 32) * v (D layout) ==========
    v8f att[2][2];
#pragma unroll
    for (int mi = 0; mi < 2; ++mi) {
#pragma unroll
        for (int m = 0; m < 8; ++m) {
            float l0 = q2[mi][0][m] * k2[mi][0][m];
            float l1 = q2[mi][1][m] * k2[mi][1][m];
            float e0 = __expf(l0), e1 = __expf(l1);
            float s = e0 + e1;
            s += __shfl_xor(s, 1, 16);
            s += __shfl_xor(s, 2, 16);
            s += __shfl_xor(s, 4, 16);
            s += __shfl_xor(s, 8, 16);
            float inv = 1.0f / s;
            att[mi][0][m] = e0 * inv * v2[mi][0][m];
            att[mi][1][m] = e1 * inv * v2[mi][1][m];
        }
    }

    // ========== attention tower: 32 -> 32 (LN,ReLU) -> 16 ==========
#pragma unroll
    for (int mi = 0; mi < 2; ++mi) {
        store_tile_h(att[mi][0], hbuf + mi * SLAB_H, 0, lane);
        store_tile_h(att[mi][1], hbuf + mi * SLAB_H, 16, lane);
    }
    LDS_FENCE();
    v16h fa[2];
#pragma unroll
    for (int mi = 0; mi < 2; ++mi) fa[mi] = afrag_lds(hbuf + mi * SLAB_H, 0, lane);
    LDS_FENCE();
    {
        v16h b[2];
        b[0] = load_bfrag(aW1p, 0, lane);
        b[1] = load_bfrag(aW1p, 1, lane);
#pragma unroll
        for (int nt = 0; nt < 2; ++nt)
#pragma unroll
            for (int mi = 0; mi < 2; ++mi) acc[mi][nt] = wmma16(fa[mi], b[nt], vzero8());
    }
    ln_transpose<2, 32>(acc, hbuf, ag1p, ab1p, lane, fr);
    v8f attOut[2];
    {
        v16h b = load_bfrag(aW2p, 0, lane);
        float bb = cst.ab2[col];
#pragma unroll
        for (int mi = 0; mi < 2; ++mi) {
            v8f a = wmma16(fr[mi][0], b, vzero8());
#pragma unroll
            for (int m = 0; m < 8; ++m) a[m] += bb;
            attOut[mi] = a;
        }
    }

    // ========== residual tail: LN(r2),ReLU -> 32 -> 16 (+rb3) ==========
#pragma unroll
    for (int mi = 0; mi < 2; ++mi) { acc[mi][0] = r2[mi][0]; acc[mi][1] = r2[mi][1]; }
    ln_transpose<2, 32>(acc, hbuf, rg2p, rb2p, lane, fr);
    v8f resOut[2];
    {
        v16h b = load_bfrag(rW3p, 0, lane);
        float bb = cst.rb3[col];
#pragma unroll
        for (int mi = 0; mi < 2; ++mi) {
            v8f a = wmma16(fr[mi][0], b, vzero8());
#pragma unroll
            for (int m = 0; m < 8; ++m) a[m] += bb;
            resOut[mi] = a;
        }
    }

    // ========== gate tower: concat(att,res)=32 -> 16 (LN,ReLU) -> 3 ==========
#pragma unroll
    for (int mi = 0; mi < 2; ++mi) {
        store_tile_h(attOut[mi], hbuf + mi * SLAB_H, 0, lane);
        store_tile_h(resOut[mi], hbuf + mi * SLAB_H, 16, lane);
    }
    LDS_FENCE();
    v16h ff[2];
#pragma unroll
    for (int mi = 0; mi < 2; ++mi) ff[mi] = afrag_lds(hbuf + mi * SLAB_H, 0, lane);
    LDS_FENCE();
    {
        v16h b = load_bfrag(gW1p, 0, lane);
#pragma unroll
        for (int mi = 0; mi < 2; ++mi) acc[mi][0] = wmma16(ff[mi], b, vzero8());
    }
    ln_transpose<1, 16>(acc, hbuf, gg1p, gb1p, lane, fr);
    {
        v16h b = load_bfrag(gW2p, 0, lane);
#pragma unroll
        for (int mi = 0; mi < 2; ++mi) {
            v8f go = wmma16(fr[mi][0], b, vzero8());
            if (col < 3) {
                float bb = cst.gb2[col];
#pragma unroll
                for (int m = 0; m < 8; ++m)
                    out[(size_t)(rowBase + mi * 16 + m + hi8) * 3 + col] = go[m] + bb;
            }
        }
    }
}

// ---------------------------------------------------------------------------
extern "C" void kernel_launch(void* const* d_in, const int* in_sizes, int n_in,
                              void* d_out, int out_size, void* d_ws, size_t ws_size,
                              hipStream_t stream) {
    (void)n_in; (void)out_size; (void)ws_size;
    const float* rep = (const float*)d_in[0];
    const float* par = (const float*)d_in[1];
    _Float16* ws = (_Float16*)d_ws;

    struct Mat { const void* W; int K, N; size_t off; };
    const Mat mats[13] = {
        { d_in[2],  128, 64,     0 },  // kW1
        { d_in[12], 128, 64,  8192 },  // vW1
        { d_in[22], 131, 64, 16384 },  // rW1
        { d_in[5],   64, 32, 26624 },  // kW2
        { d_in[15],  64, 32, 28672 },  // vW2
        { d_in[25],  64, 32, 30720 },  // rW2
        { d_in[7],    3, 16, 32768 },  // qW1
        { d_in[10],  16, 32, 33280 },  // qW2
        { d_in[17],  32, 32, 34304 },  // aW1
        { d_in[20],  32, 16, 35328 },  // aW2
        { d_in[28],  32, 16, 35840 },  // rW3
        { d_in[30],  32, 16, 36352 },  // gW1
        { d_in[33],  16,  3, 36864 },  // gW2
    };
    for (int i = 0; i < 13; ++i) {
        int Kt = (mats[i].K + 31) / 32;
        int Nt = (mats[i].N + 15) / 16;
        int total = Kt * Nt * 512;
        pack_w_kernel<<<(total + 255) / 256, 256, 0, stream>>>(
            (const float*)mats[i].W, ws + mats[i].off, mats[i].K, mats[i].N, Nt, total);
    }

    struct Vec { const void* v; int dim; size_t off; };
    const Vec vecs[14] = {
        { d_in[3],  64, 37376 },  // kg1
        { d_in[4],  64, 38400 },  // kb1
        { d_in[13], 64, 39424 },  // vg1
        { d_in[14], 64, 40448 },  // vb1
        { d_in[23], 64, 41472 },  // rg1
        { d_in[24], 64, 42496 },  // rb1
        { d_in[26], 32, 43520 },  // rg2
        { d_in[27], 32, 44032 },  // rb2
        { d_in[18], 32, 44544 },  // ag1
        { d_in[19], 32, 45056 },  // ab1
        { d_in[8],  16, 45568 },  // qg1
        { d_in[9],  16, 46080 },  // qb1
        { d_in[31], 16, 46592 },  // gg1
        { d_in[32], 16, 47104 },  // gb1
    };
    for (int i = 0; i < 14; ++i) {
        int kt = (vecs[i].dim + 31) / 32;
        int total = kt * 512;
        pack_vec_kernel<<<(total + 255) / 256, 256, 0, stream>>>(
            (const float*)vecs[i].v, ws + vecs[i].off, vecs[i].dim, total);
    }

    TailConsts cst = {
        (const float*)d_in[6],   // kb2
        (const float*)d_in[16],  // vb2
        (const float*)d_in[11],  // qb2
        (const float*)d_in[21],  // ab2
        (const float*)d_in[29],  // rb3
        (const float*)d_in[34],  // gb2
    };

    int rows = in_sizes[0] / 128;     // N = 1,048,576
    int grid = rows >> 8;             // 256 rows per block (8 waves x 32 rows)
    fused_towers_kernel<<<grid, 256, 0, stream>>>(rep, par, ws, cst, (float*)d_out);
}
